// MultiHeadAttention_36842229465921
// MI455X (gfx1250) — matrix operand order repaired and checked by reading, not compiled, not run on hardware
//
#include <hip/hip_runtime.h>
#include <stdint.h>

#define NB   8
#define SEQ  2048
#define EMB  512
#define NH   8
#define DH   64
#define LN_EPS 1e-5f

typedef __attribute__((ext_vector_type(16))) __bf16 v16bf;
typedef __attribute__((ext_vector_type(8)))  float  v8f;

union FragB16 { v16bf v; uint4 q[2]; };

__device__ __forceinline__ unsigned short f2bf(float f) {
  union { float f; unsigned u; } cv; cv.f = f;
  unsigned u = cv.u;
  u += 0x7FFFu + ((u >> 16) & 1u);   // round-to-nearest-even
  return (unsigned short)(u >> 16);
}

// A-fragment: 16x32 row-major tile at (rows base, k0); ld in elements.
// ISA layout: lane<16 -> row=lane, K = k0+[0..7] and k0+[16..23]
//             lane>=16 -> row=lane-16, K = k0+[8..15] and k0+[24..31]
__device__ __forceinline__ FragB16 load_frag_a(const unsigned short* base, int ld, int k0) {
  int lane = threadIdx.x & 31;
  int r = lane & 15, half = lane >> 4;
  const unsigned short* p = base + r * ld + k0 + half * 8;
  FragB16 f;
  f.q[0] = *(const uint4*)(p);
  f.q[1] = *(const uint4*)(p + 16);
  return f;
}

// B-fragment from "BT" storage: row n of BT = column n of B (K contiguous).
// Layout: lanes 0-15 hold K=k0+[0..7],[16..23]; lanes 16-31 hold K=k0+[8..15],[24..31].
__device__ __forceinline__ FragB16 load_frag_b(const unsigned short* bt, int ld, int k0) {
  int lane = threadIdx.x & 31;
  int n = lane & 15, half = lane >> 4;
  const unsigned short* p = bt + n * ld + k0 + half * 8;
  FragB16 f;
  f.q[0] = *(const uint4*)(p);
  f.q[1] = *(const uint4*)(p + 16);
  return f;
}

__device__ __forceinline__ v8f wmma_bf16(const FragB16& a, const FragB16& b, v8f c) {
  return __builtin_amdgcn_wmma_f32_16x16x32_bf16(false, a.v, false, b.v, (short)0, c, false, false);
}

// ---------------- prep kernels ----------------

__global__ void cvt_f32_bf16(const float* __restrict__ src, unsigned short* __restrict__ dst, int n) {
  int i = blockIdx.x * blockDim.x + threadIdx.x;
  if (i < n) dst[i] = f2bf(src[i]);
}

// WT[h][n][k] = W[h][k][n]  (bf16)
__global__ void transpose_w(const float* __restrict__ W, unsigned short* __restrict__ WT,
                            int H, int K, int N) {
  int i = blockIdx.x * blockDim.x + threadIdx.x;
  int total = H * K * N;
  if (i >= total) return;
  int h = i / (K * N);
  int rem = i - h * K * N;
  int k = rem / N;
  int n = rem - k * N;
  WT[(h * N + n) * K + k] = f2bf(W[i]);
}

// ---------------- QKV projection: out = x @ W[h] + bias[h] ----------------
// vtrans==0: out[b,h,s,d] row-major (Q, K)
// vtrans==1: out[b,h,d,s] transposed (V) -> contiguous b128 stores, and the
//            attention P@V B-fragments read it directly from global.

__global__ void __launch_bounds__(128) qkv_gemm(const unsigned short* __restrict__ xh, // [NB,SEQ,EMB]
                                                const unsigned short* __restrict__ WT, // [NH,DH,EMB]
                                                const float* __restrict__ bias,        // [NH,DH]
                                                unsigned short* __restrict__ out,
                                                int vtrans) {
  const int S16 = SEQ / 16;
  int wave = threadIdx.x >> 5;
  int tile = blockIdx.x * 4 + wave;
  int b = tile / (NH * S16);
  int rem = tile - b * NH * S16;
  int h = rem / S16;
  int s0 = (rem - h * S16) * 16;
  int lane = threadIdx.x & 31;
  int col = lane & 15, half = lane >> 4;

  const unsigned short* abase = xh + ((size_t)b * SEQ + s0) * EMB;
  const unsigned short* wbase = WT + (size_t)h * DH * EMB;

  v8f acc[4] = {};
  for (int k0 = 0; k0 < EMB; k0 += 32) {
    FragB16 a = load_frag_a(abase, EMB, k0);
#pragma unroll
    for (int nt = 0; nt < 4; ++nt) {
      FragB16 bf = load_frag_b(wbase + nt * 16 * EMB, EMB, k0);
      acc[nt] = wmma_bf16(a, bf, acc[nt]);
    }
  }
  if (vtrans) {
    // out[b,h,d,s]: lane owns column d=n -> 8 consecutive s values per frag
    unsigned short* obase = out + (((size_t)b * NH + h) * DH) * SEQ;
#pragma unroll
    for (int nt = 0; nt < 4; ++nt) {
      int n = nt * 16 + col;
      float bv = bias[h * DH + n];
      unsigned short pk[8];
#pragma unroll
      for (int r = 0; r < 8; ++r) pk[r] = f2bf(acc[nt][r] + bv);
      *(uint4*)(obase + (size_t)n * SEQ + s0 + 8 * half) = *(const uint4*)pk;
    }
  } else {
    unsigned short* obase = out + (((size_t)b * NH + h) * SEQ + s0) * DH;
#pragma unroll
    for (int nt = 0; nt < 4; ++nt) {
      int n = nt * 16 + col;
      float bv = bias[h * DH + n];
#pragma unroll
      for (int r = 0; r < 8; ++r) {
        int m = r + 8 * half;
        obase[m * DH + n] = f2bf(acc[nt][r] + bv);
      }
    }
  }
}

// ---------------- flash attention: one wave per 16 query rows of one (b,h) ----------------
// K chunks are double-buffered into LDS with global_load_async_to_lds_b128
// (ASYNCcnt), overlapping the DMA of chunk i+1 with the WMMA work of chunk i.

__global__ void __launch_bounds__(32) attention(const unsigned short* __restrict__ Qh, // [NB,NH,SEQ,DH]
                                                const unsigned short* __restrict__ Kh, // [NB,NH,SEQ,DH]
                                                const unsigned short* __restrict__ Vt, // [NB,NH,DH,SEQ]
                                                unsigned short* __restrict__ concat) { // [NB,SEQ,EMB]
  __shared__ unsigned short Pl[16 * 32];        // probabilities, row-major bf16
  __shared__ unsigned short Kbuf[2][32 * DH];   // double-buffered K chunk (2 x 4KB)
  const int S16 = SEQ / 16;
  int tile = blockIdx.x;
  int b = tile / (NH * S16);
  int rem = tile - b * NH * S16;
  int h = rem / S16;
  int s0 = (rem - h * S16) * 16;
  int lane = threadIdx.x & 31;
  int col = lane & 15, half = lane >> 4;

  const unsigned short* qbase = Qh + (((size_t)b * NH + h) * SEQ + s0) * DH;
  const unsigned short* kbase = Kh + (((size_t)b * NH + h) * SEQ) * DH;
  const unsigned short* vbase = Vt + (((size_t)b * NH + h) * DH) * SEQ;

  // async-stage one 32-key K chunk (32 rows x 128B) into LDS buffer `buf`
  auto stage_k = [&](int t0s, int buf) {
    const unsigned short* src = kbase + (size_t)(t0s + lane) * DH;
    unsigned short* dst = &Kbuf[buf][lane * DH];
#pragma unroll
    for (int j = 0; j < 8; ++j) {
      unsigned lds_off = (unsigned)(uintptr_t)(dst + j * 8);
      unsigned long long ga = (unsigned long long)(uintptr_t)(src + j * 8);
      asm volatile("global_load_async_to_lds_b128 %0, %1, off"
                   :: "v"(lds_off), "v"(ga) : "memory");
    }
  };

  FragB16 qf0 = load_frag_a(qbase, DH, 0);
  FragB16 qf1 = load_frag_a(qbase, DH, 32);

  float mrow[8], lrow[8];
#pragma unroll
  for (int r = 0; r < 8; ++r) { mrow[r] = -3.0e38f; lrow[r] = 0.0f; }
  v8f acc[4] = {};
  const float scale = 0.125f;  // 1/sqrt(64)

  stage_k(0, 0);

  for (int t0 = 0, it = 0; t0 < SEQ; t0 += 32, ++it) {
    int cur = it & 1;
    if (t0 + 32 < SEQ) {
      stage_k(t0 + 32, cur ^ 1);
      asm volatile("s_wait_asynccnt 0x8" ::: "memory");  // oldest 8 (current chunk) done
    } else {
      asm volatile("s_wait_asynccnt 0x0" ::: "memory");
    }

    // S = Q @ K^T for this key chunk (two 16-key tiles, d = 64 = 2 wmma k-steps)
    const unsigned short* kc = &Kbuf[cur][0];
    v8f z = {};
    FragB16 kb;
    kb = load_frag_b(kc, DH, 0);
    v8f sA = wmma_bf16(qf0, kb, z);
    kb = load_frag_b(kc, DH, 32);
    sA = wmma_bf16(qf1, kb, sA);
    kb = load_frag_b(kc + 16 * DH, DH, 0);
    v8f sB = wmma_bf16(qf0, kb, z);
    kb = load_frag_b(kc + 16 * DH, DH, 32);
    sB = wmma_bf16(qf1, kb, sB);

    // online softmax update (row = r + 8*half, columns striped over 16-lane groups)
#pragma unroll
    for (int r = 0; r < 8; ++r) {
      float a0 = sA[r] * scale;
      float a1 = sB[r] * scale;
      float mx = fmaxf(a0, a1);
#pragma unroll
      for (int off = 1; off < 16; off <<= 1)
        mx = fmaxf(mx, __shfl_xor(mx, off, 32));
      float mnew = fmaxf(mrow[r], mx);
      float alpha = __expf(mrow[r] - mnew);
      float p0 = __expf(a0 - mnew);
      float p1 = __expf(a1 - mnew);
      float rs = p0 + p1;
#pragma unroll
      for (int off = 1; off < 16; off <<= 1)
        rs += __shfl_xor(rs, off, 32);
      lrow[r] = lrow[r] * alpha + rs;
      mrow[r] = mnew;
      acc[0][r] *= alpha; acc[1][r] *= alpha; acc[2][r] *= alpha; acc[3][r] *= alpha;
      int m = r + 8 * half;
      Pl[m * 32 + col]      = f2bf(p0);
      Pl[m * 32 + 16 + col] = f2bf(p1);
    }
    __syncthreads();

    // ctx += P @ V: A-frag of P from LDS, B-frags straight from transposed V in global
    FragB16 pf;
    {
      const unsigned short* p = &Pl[col * 32 + half * 8];
      pf.q[0] = *(const uint4*)(p);
      pf.q[1] = *(const uint4*)(p + 16);
    }
#pragma unroll
    for (int nt = 0; nt < 4; ++nt) {
      FragB16 vf;
      const unsigned short* p = vbase + (size_t)(nt * 16 + col) * SEQ + t0 + 8 * half;
      vf.q[0] = *(const uint4*)(p);
      vf.q[1] = *(const uint4*)(p + 16);
      acc[nt] = wmma_bf16(pf, vf, acc[nt]);
    }
    __syncthreads();
  }

  // normalize and write concat[b, s, h*64 + n] (bf16)
  unsigned short* obase = concat + ((size_t)b * SEQ + s0) * EMB + h * DH;
#pragma unroll
  for (int r = 0; r < 8; ++r) {
    float inv = 1.0f / lrow[r];
    int m = r + 8 * half;
#pragma unroll
    for (int nt = 0; nt < 4; ++nt)
      obase[(size_t)m * EMB + nt * 16 + col] = f2bf(acc[nt][r] * inv);
  }
}

// ---------------- output projection + residual + LayerNorm (fused) ----------------

__global__ void __launch_bounds__(256) outproj_ln(const unsigned short* __restrict__ concat, // [NB*SEQ,EMB]
                                                  const unsigned short* __restrict__ WoT,    // [EMB,EMB]
                                                  const float* __restrict__ bo,
                                                  const float* __restrict__ x,               // residual
                                                  const float* __restrict__ gamma,
                                                  const float* __restrict__ beta,
                                                  float* __restrict__ out) {
  __shared__ float Y[16][EMB];  // 32KB of the 320KB WGP LDS
  int m0 = blockIdx.x * 16;
  int wave = threadIdx.x >> 5;
  int lane = threadIdx.x & 31;
  int col = lane & 15, half = lane >> 4;

  const unsigned short* abase = concat + (size_t)m0 * EMB;
  v8f acc[4] = {};
  for (int k0 = 0; k0 < EMB; k0 += 32) {
    FragB16 a = load_frag_a(abase, EMB, k0);
#pragma unroll
    for (int nt = 0; nt < 4; ++nt) {
      const unsigned short* bt = WoT + (size_t)(wave * 64 + nt * 16) * EMB;
      FragB16 bf = load_frag_b(bt, EMB, k0);
      acc[nt] = wmma_bf16(a, bf, acc[nt]);
    }
  }
#pragma unroll
  for (int nt = 0; nt < 4; ++nt) {
    int n = wave * 64 + nt * 16 + col;
    float bias = bo[n];
#pragma unroll
    for (int r = 0; r < 8; ++r)
      Y[r + 8 * half][n] = acc[nt][r] + bias;
  }
  __syncthreads();

  // LayerNorm: each half-wave handles one row (16 lanes x 32 columns each)
  int row = 2 * wave + half;
  float sum = 0.f, sq = 0.f;
  float yv[32];
#pragma unroll
  for (int j = 0; j < 32; ++j) {
    int c = col + 16 * j;
    float v = Y[row][c] + x[(size_t)(m0 + row) * EMB + c];
    yv[j] = v;
    sum += v; sq += v * v;
  }
#pragma unroll
  for (int off = 1; off < 16; off <<= 1) {
    sum += __shfl_xor(sum, off, 32);
    sq  += __shfl_xor(sq,  off, 32);
  }
  float mean = sum * (1.0f / EMB);
  float var  = sq * (1.0f / EMB) - mean * mean;
  float rstd = rsqrtf(var + LN_EPS);
#pragma unroll
  for (int j = 0; j < 32; ++j) {
    int c = col + 16 * j;
    out[(size_t)(m0 + row) * EMB + c] = (yv[j] - mean) * rstd * gamma[c] + beta[c];
  }
}

// ---------------- launch ----------------

extern "C" void kernel_launch(void* const* d_in, const int* in_sizes, int n_in,
                              void* d_out, int out_size, void* d_ws, size_t ws_size,
                              hipStream_t stream) {
  (void)in_sizes; (void)n_in; (void)out_size; (void)ws_size;
  const float* x     = (const float*)d_in[0];
  const float* Wq    = (const float*)d_in[1];
  const float* Wk    = (const float*)d_in[2];
  const float* Wv    = (const float*)d_in[3];
  const float* bq    = (const float*)d_in[4];
  const float* bk    = (const float*)d_in[5];
  const float* bv    = (const float*)d_in[6];
  const float* Wo    = (const float*)d_in[7];
  const float* bo    = (const float*)d_in[8];
  const float* gamma = (const float*)d_in[9];
  const float* beta  = (const float*)d_in[10];
  float* out = (float*)d_out;

  char* ws = (char*)d_ws;
  auto alloc = [&](size_t bytes) {
    char* p = ws;
    ws += (bytes + 255) & ~(size_t)255;
    return p;
  };
  unsigned short* xh  = (unsigned short*)alloc((size_t)NB * SEQ * EMB * 2);
  unsigned short* WqT = (unsigned short*)alloc((size_t)NH * DH * EMB * 2);
  unsigned short* WkT = (unsigned short*)alloc((size_t)NH * DH * EMB * 2);
  unsigned short* WvT = (unsigned short*)alloc((size_t)NH * DH * EMB * 2);
  unsigned short* WoT = (unsigned short*)alloc((size_t)EMB * EMB * 2);
  unsigned short* Qh  = (unsigned short*)alloc((size_t)NB * NH * SEQ * DH * 2);
  unsigned short* Kh  = (unsigned short*)alloc((size_t)NB * NH * SEQ * DH * 2);
  unsigned short* Vtr = (unsigned short*)alloc((size_t)NB * NH * DH * SEQ * 2);
  unsigned short* cat = (unsigned short*)alloc((size_t)NB * SEQ * EMB * 2);

  // prep: bf16 conversions + weight transposes
  int n = NB * SEQ * EMB;
  cvt_f32_bf16<<<(n + 255) / 256, 256, 0, stream>>>(x, xh, n);
  int nw = NH * EMB * DH;
  transpose_w<<<(nw + 255) / 256, 256, 0, stream>>>(Wq, WqT, NH, EMB, DH);
  transpose_w<<<(nw + 255) / 256, 256, 0, stream>>>(Wk, WkT, NH, EMB, DH);
  transpose_w<<<(nw + 255) / 256, 256, 0, stream>>>(Wv, WvT, NH, EMB, DH);
  int nwo = EMB * EMB;
  transpose_w<<<(nwo + 255) / 256, 256, 0, stream>>>(Wo, WoT, 1, EMB, EMB);

  // QKV projections (WMMA); V stored transposed [b,h,d,s]
  int tiles = NB * NH * (SEQ / 16);  // 8192
  qkv_gemm<<<tiles / 4, 128, 0, stream>>>(xh, WqT, bq, Qh, 0);
  qkv_gemm<<<tiles / 4, 128, 0, stream>>>(xh, WkT, bk, Kh, 0);
  qkv_gemm<<<tiles / 4, 128, 0, stream>>>(xh, WvT, bv, Vtr, 1);

  // flash attention (WMMA + async-to-LDS K streaming), writes concat layout
  attention<<<tiles, 32, 0, stream>>>(Qh, Kh, Vtr, cat);

  // output projection + residual + LayerNorm (WMMA + fused epilogue)
  outproj_ln<<<(NB * SEQ) / 16, 256, 0, stream>>>(cat, WoT, bo, x, gamma, beta, out);
}
